// Rwkv_Tmix_x070_68075231642002
// MI455X (gfx1250) — matrix operand order repaired and checked by reading, not compiled, not run on hardware
//
#include <hip/hip_runtime.h>
#include <math.h>

// ---------------- problem constants ----------------
#define Bn 2
#define Tn 1024
#define Cn 2048
#define Hh 32
#define Nn 64
#define BT (Bn*Tn)
#define GN_EPS (1e-5f * 64.0f)

// ---------------- WMMA types ----------------
typedef __attribute__((ext_vector_type(16))) __bf16 v16bf;
typedef __attribute__((ext_vector_type(8)))  float  v8f;

union V16 { uint4 q[2]; v16bf v; };

// ---------------- CDNA5 async global->LDS path (probe-gated) ----------------
#if defined(__gfx1250__) && \
    __has_builtin(__builtin_amdgcn_global_load_async_to_lds_b128) && \
    __has_builtin(__builtin_amdgcn_s_wait_asynccnt)
#define HAVE_ASYNC 1
#else
#define HAVE_ASYNC 0
#endif

#if HAVE_ASYNC
typedef int v4i_t __attribute__((vector_size(16)));
typedef __attribute__((address_space(1))) v4i_t* gv4i_p;   // global src
typedef __attribute__((address_space(3))) v4i_t* lv4i_p;   // LDS dst

__device__ __forceinline__ void async_cp16(const __bf16* g, __bf16* l) {
    void* gv = (void*)g;   // drop const
    void* lv = (void*)l;
    __builtin_amdgcn_global_load_async_to_lds_b128((gv4i_p)gv, (lv4i_p)lv, 0, 0);
}
#endif

// =====================================================================
// fp32 -> bf16 conversion
// =====================================================================
__global__ void k_cvt_bf16(const float* __restrict__ src, __bf16* __restrict__ dst, size_t n) {
    size_t i = (size_t)blockIdx.x * blockDim.x + threadIdx.x;
    if (i < n) dst[i] = (__bf16)src[i];
}

// =====================================================================
// token shift + six mixes, emitted as bf16 GEMM inputs
// =====================================================================
__global__ void k_mix(const float* __restrict__ h, const float* __restrict__ shift,
                      const float* __restrict__ mr, const float* __restrict__ mw,
                      const float* __restrict__ mk, const float* __restrict__ mv,
                      const float* __restrict__ ma, const float* __restrict__ mg,
                      __bf16* __restrict__ oxr, __bf16* __restrict__ oxw,
                      __bf16* __restrict__ oxk, __bf16* __restrict__ oxv,
                      __bf16* __restrict__ oxa, __bf16* __restrict__ oxg) {
    size_t idx = (size_t)blockIdx.x * blockDim.x + threadIdx.x;
    size_t total = (size_t)BT * Cn;
    if (idx >= total) return;
    int c = (int)(idx % Cn);
    size_t bt = idx / Cn;
    int t = (int)(bt % Tn);
    int b = (int)(bt / Tn);
    float cur  = h[idx];
    float prev = (t == 0) ? shift[(size_t)b * Cn + c] : h[idx - Cn];
    float xx = prev - cur;
    oxr[idx] = (__bf16)(cur + xx * mr[c]);
    oxw[idx] = (__bf16)(cur + xx * mw[c]);
    oxk[idx] = (__bf16)(cur + xx * mk[c]);
    oxv[idx] = (__bf16)(cur + xx * mv[c]);
    oxa[idx] = (__bf16)(cur + xx * ma[c]);
    oxg[idx] = (__bf16)(cur + xx * mg[c]);
}

// =====================================================================
// bf16 WMMA GEMM: C[M,N] = A[M,K] @ W[N,K]^T   (both K-contiguous)
// block = 256 threads (8 waves), tile 128x128, KB = 32
// wave grid 4(M) x 2(N); each wave: 2x4 of 16x16 WMMA accum tiles
// Async path: double-buffered LDS, GLOBAL_LOAD_ASYNC_TO_LDS_B128 prefetch,
// one barrier per K-step. Sync fallback: classic 2-barrier staging.
// =====================================================================
__global__ __launch_bounds__(256)
void k_gemm_bf16(const __bf16* __restrict__ A, const __bf16* __restrict__ W,
                 float* __restrict__ C, int M, int N, int K) {
    __shared__ __bf16 As[2][128][40];   // +8 pad keeps 16B alignment, kills conflicts
    __shared__ __bf16 Bs[2][128][40];

    const int tid  = threadIdx.x;
    const int wave = tid >> 5;
    const int lane = tid & 31;
    const int half = lane >> 4;      // 0: lanes 0-15, 1: lanes 16-31
    const int l16  = lane & 15;
    const int wm   = wave >> 1;      // 0..3  -> M offset wm*32
    const int wn   = wave & 1;       // 0..1  -> N offset wn*64
    const int gm0  = blockIdx.y * 128;
    const int gn0  = blockIdx.x * 128;
    const int lr   = tid >> 1;           // tile row this thread loads
    const int lcs  = (tid & 1) * 16;     // col segment (16 bf16 = 32B)

    const __bf16* gA = A + (size_t)(gm0 + lr) * K + lcs;
    const __bf16* gB = W + (size_t)(gn0 + lr) * K + lcs;

    v8f acc[2][4];
    #pragma unroll
    for (int i = 0; i < 2; ++i)
        #pragma unroll
        for (int j = 0; j < 4; ++j)
            acc[i][j] = v8f{0.f,0.f,0.f,0.f,0.f,0.f,0.f,0.f};

#if HAVE_ASYNC
    // prologue: prefetch tile 0 into buffer 0
    async_cp16(gA,     &As[0][lr][lcs]);
    async_cp16(gA + 8, &As[0][lr][lcs + 8]);
    async_cp16(gB,     &Bs[0][lr][lcs]);
    async_cp16(gB + 8, &Bs[0][lr][lcs + 8]);

    int buf = 0;
    for (int kt = 0; kt < K; kt += 32, buf ^= 1) {
        __builtin_amdgcn_s_wait_asynccnt(0);
        __syncthreads();                  // tile ready; other buffer idle
        if (kt + 32 < K) {                // prefetch next tile into idle buffer
            const __bf16* na = gA + kt + 32;
            const __bf16* nb = gB + kt + 32;
            async_cp16(na,     &As[buf ^ 1][lr][lcs]);
            async_cp16(na + 8, &As[buf ^ 1][lr][lcs + 8]);
            async_cp16(nb,     &Bs[buf ^ 1][lr][lcs]);
            async_cp16(nb + 8, &Bs[buf ^ 1][lr][lcs + 8]);
        }
        V16 fa[2], fb[4];
        #pragma unroll
        for (int i = 0; i < 2; ++i) {
            int m = wm * 32 + i * 16 + l16;
            fa[i].q[0] = *(const uint4*)&As[buf][m][half * 8];
            fa[i].q[1] = *(const uint4*)&As[buf][m][16 + half * 8];
        }
        #pragma unroll
        for (int j = 0; j < 4; ++j) {
            int n = wn * 64 + j * 16 + l16;
            fb[j].q[0] = *(const uint4*)&Bs[buf][n][half * 8];
            fb[j].q[1] = *(const uint4*)&Bs[buf][n][16 + half * 8];
        }
        #pragma unroll
        for (int i = 0; i < 2; ++i)
            #pragma unroll
            for (int j = 0; j < 4; ++j)
                acc[i][j] = __builtin_amdgcn_wmma_f32_16x16x32_bf16(
                    false, fa[i].v, false, fb[j].v, (short)0, acc[i][j], false, false);
    }
#else
    for (int kt = 0; kt < K; kt += 32) {
        const uint4* ga = (const uint4*)(gA + kt);
        const uint4* gb = (const uint4*)(gB + kt);
        uint4 a0 = ga[0], a1 = ga[1];
        uint4 b0 = gb[0], b1 = gb[1];
        *(uint4*)&As[0][lr][lcs]     = a0;
        *(uint4*)&As[0][lr][lcs + 8] = a1;
        *(uint4*)&Bs[0][lr][lcs]     = b0;
        *(uint4*)&Bs[0][lr][lcs + 8] = b1;
        __syncthreads();

        V16 fa[2], fb[4];
        #pragma unroll
        for (int i = 0; i < 2; ++i) {
            int m = wm * 32 + i * 16 + l16;
            fa[i].q[0] = *(const uint4*)&As[0][m][half * 8];
            fa[i].q[1] = *(const uint4*)&As[0][m][16 + half * 8];
        }
        #pragma unroll
        for (int j = 0; j < 4; ++j) {
            int n = wn * 64 + j * 16 + l16;
            fb[j].q[0] = *(const uint4*)&Bs[0][n][half * 8];
            fb[j].q[1] = *(const uint4*)&Bs[0][n][16 + half * 8];
        }
        #pragma unroll
        for (int i = 0; i < 2; ++i)
            #pragma unroll
            for (int j = 0; j < 4; ++j)
                acc[i][j] = __builtin_amdgcn_wmma_f32_16x16x32_bf16(
                    false, fa[i].v, false, fb[j].v, (short)0, acc[i][j], false, false);
        __syncthreads();
    }
#endif

    // C/D layout: vgpr r holds m = r + 8*half, n = l16
    #pragma unroll
    for (int i = 0; i < 2; ++i) {
        #pragma unroll
        for (int j = 0; j < 4; ++j) {
            int n = gn0 + wn * 64 + j * 16 + l16;
            #pragma unroll
            for (int r = 0; r < 8; ++r) {
                int m = gm0 + wm * 32 + i * 16 + half * 8 + r;
                C[(size_t)m * N + n] = acc[i][j][r];
            }
        }
    }
}

// =====================================================================
// small low-rank GEMM stage 1: out[M,Nsm] = act(A_bf16[M,K] @ Wsm[K,Nsm])
// act: 0 none, 1 tanh, 2 sigmoid
// =====================================================================
__global__ void k_lr1(const __bf16* __restrict__ A, const float* __restrict__ Wsm,
                      float* __restrict__ out, int M, int K, int Nsm, int act) {
    size_t idx = (size_t)blockIdx.x * blockDim.x + threadIdx.x;
    if (idx >= (size_t)M * Nsm) return;
    int m = (int)(idx / Nsm), n = (int)(idx % Nsm);
    const __bf16* arow = A + (size_t)m * K;
    float s = 0.f;
    for (int k = 0; k < K; ++k) s += (float)arow[k] * Wsm[(size_t)k * Nsm + n];
    if (act == 1) s = tanhf(s);
    else if (act == 2) s = 1.f / (1.f + expf(-s));
    out[idx] = s;
}

// =====================================================================
// fused stage 2 + elementwise (see round-1 comments)
// =====================================================================
__global__ __launch_bounds__(256)
void k_fuse(const float* __restrict__ hw, const float* __restrict__ ha,
            const float* __restrict__ hv, const float* __restrict__ hg,
            const float* __restrict__ w0, const float* __restrict__ a0,
            const float* __restrict__ v0,
            const float* __restrict__ w2, const float* __restrict__ a2,
            const float* __restrict__ v2, const float* __restrict__ g2,
            const float* __restrict__ k_k, const float* __restrict__ k_a,
            const float* __restrict__ v_first,
            float* __restrict__ k_io, float* __restrict__ v_io,
            float* __restrict__ dec, float* __restrict__ aa,
            float* __restrict__ bb, float* __restrict__ g_out) {
    int lane = threadIdx.x & 31;
    int row  = blockIdx.x * 8 + (threadIdx.x >> 5);
    if (row >= BT * Hh) return;
    int m = row / Hh, h = row % Hh;

    float kkv[2], ahv[2];
    float sumsq = 0.f;
    #pragma unroll
    for (int p = 0; p < 2; ++p) {
        int n = lane + p * 32;
        int c = h * Nn + n;
        size_t idx = (size_t)m * Cn + c;

        float lw = w0[c];
        for (int j = 0; j < 64; ++j)  lw += hw[(size_t)m * 64 + j]  * w2[(size_t)j * Cn + c];
        float w = -log1pf(expf(-lw)) - 0.5f;
        dec[idx] = expf(w);

        float av = a0[c];
        for (int j = 0; j < 64; ++j)  av += ha[(size_t)m * 64 + j]  * a2[(size_t)j * Cn + c];
        float ah = 1.f / (1.f + expf(-av));

        float vv = v0[c];
        for (int j = 0; j < 32; ++j)  vv += hv[(size_t)m * 32 + j]  * v2[(size_t)j * Cn + c];
        float vs = 1.f / (1.f + expf(-vv));

        float gv = 0.f;
        for (int j = 0; j < 128; ++j) gv += hg[(size_t)m * 128 + j] * g2[(size_t)j * Cn + c];
        g_out[idx] = gv;

        float kraw = k_io[idx];
        float vraw = v_io[idx];
        v_io[idx] = vraw + (v_first[idx] - vraw) * vs;
        float kk = kraw * k_k[c];
        kkv[p] = kk; ahv[p] = ah;
        k_io[idx] = kraw * (1.f + k_a[c] * (ah - 1.f));
        sumsq += kk * kk;
    }
    #pragma unroll
    for (int msk = 16; msk >= 1; msk >>= 1) sumsq += __shfl_xor(sumsq, msk, 32);
    float inv = 1.f / fmaxf(sqrtf(sumsq), 1e-12f);
    #pragma unroll
    for (int p = 0; p < 2; ++p) {
        int n = lane + p * 32;
        size_t idx = (size_t)m * Cn + h * Nn + n;
        float kkn = kkv[p] * inv;
        aa[idx] = -kkn;
        bb[idx] = kkn * ahv[p];
    }
}

// =====================================================================
// WKV7 scan: one block per (b,h); 64 threads; thread tid owns state row
// S[v=tid][k=0..63] in registers; per-step vectors staged in LDS in
// chunks of TC steps; next chunk prefetched (global_prefetch_b8).
// =====================================================================
#define TC 8
__global__ __launch_bounds__(64)
void k_scan(const float* __restrict__ r, const float* __restrict__ k,
            const float* __restrict__ v, const float* __restrict__ dec,
            const float* __restrict__ aa, const float* __restrict__ bb,
            const float* __restrict__ s0, float* __restrict__ o) {
    int bh = blockIdx.x;
    int b = bh / Hh, h = bh % Hh;
    int tid = threadIdx.x;             // v index
    __shared__ float L[TC][6][Nn];

    float S[Nn];
    const float* sb = s0 + (((size_t)b * Hh + h) * Nn + tid) * Nn;
    #pragma unroll
    for (int kk = 0; kk < Nn; ++kk) S[kk] = sb[kk];

    for (int t0 = 0; t0 < Tn; t0 += TC) {
        __syncthreads();
        for (int s = 0; s < TC; ++s) {
            size_t base = (((size_t)b * Tn + t0 + s) * Hh + h) * Nn;
            L[s][0][tid] = r  [base + tid];
            L[s][1][tid] = k  [base + tid];
            L[s][2][tid] = v  [base + tid];
            L[s][3][tid] = dec[base + tid];
            L[s][4][tid] = aa [base + tid];
            L[s][5][tid] = bb [base + tid];
        }
        __syncthreads();
        if (t0 + TC < Tn) {            // hint next chunk into cache
            size_t nb = (((size_t)b * Tn + t0 + TC) * Hh + h) * Nn + tid;
            __builtin_prefetch(r   + nb, 0, 1);
            __builtin_prefetch(k   + nb, 0, 1);
            __builtin_prefetch(v   + nb, 0, 1);
            __builtin_prefetch(dec + nb, 0, 1);
            __builtin_prefetch(aa  + nb, 0, 1);
            __builtin_prefetch(bb  + nb, 0, 1);
        }
        for (int s = 0; s < TC; ++s) {
            float sa = 0.f;
            #pragma unroll
            for (int kk = 0; kk < Nn; ++kk) sa += S[kk] * L[s][4][kk];
            float vv = L[s][2][tid];
            float oacc = 0.f;
            #pragma unroll
            for (int kk = 0; kk < Nn; ++kk) {
                float Sn = S[kk] * L[s][3][kk] + sa * L[s][5][kk] + vv * L[s][1][kk];
                S[kk] = Sn;
                oacc += Sn * L[s][0][kk];
            }
            size_t base = (((size_t)b * Tn + t0 + s) * Hh + h) * Nn;
            o[base + tid] = oacc;
        }
    }
}

// =====================================================================
// groupnorm over N=64 per (b,t,h) + r*k*r_k bonus + gate g; emit bf16 y
// =====================================================================
__global__ __launch_bounds__(256)
void k_gn(const float* __restrict__ o, const float* __restrict__ r,
          const float* __restrict__ k, const float* __restrict__ v,
          const float* __restrict__ g, const float* __restrict__ gn_w,
          const float* __restrict__ gn_b, const float* __restrict__ r_k,
          __bf16* __restrict__ ybf) {
    int lane = threadIdx.x & 31;
    int row  = blockIdx.x * 8 + (threadIdx.x >> 5);
    if (row >= BT * Hh) return;
    int h = row % Hh;
    size_t base = (size_t)row * Nn;

    float x0 = o[base + lane], x1 = o[base + lane + 32];
    float s  = x0 + x1;
    float ss = x0 * x0 + x1 * x1;
    float ws = r[base + lane]      * k[base + lane]      * r_k[h * Nn + lane]
             + r[base + lane + 32] * k[base + lane + 32] * r_k[h * Nn + lane + 32];
    #pragma unroll
    for (int msk = 16; msk >= 1; msk >>= 1) {
        s  += __shfl_xor(s,  msk, 32);
        ss += __shfl_xor(ss, msk, 32);
        ws += __shfl_xor(ws, msk, 32);
    }
    float mu   = s * (1.f / Nn);
    float var  = ss * (1.f / Nn) - mu * mu;
    float rstd = rsqrtf(var + GN_EPS);
    #pragma unroll
    for (int p = 0; p < 2; ++p) {
        int n = lane + p * 32;
        int c = h * Nn + n;
        float xn = (o[base + n] - mu) * rstd * gn_w[c] + gn_b[c];
        float y  = (xn + ws * v[base + n]) * g[base + n];
        ybf[base + n] = (__bf16)y;
    }
}

// =====================================================================
// launcher
// =====================================================================
extern "C" void kernel_launch(void* const* d_in, const int* in_sizes, int n_in,
                              void* d_out, int out_size, void* d_ws, size_t ws_size,
                              hipStream_t stream) {
    const float* hid     = (const float*)d_in[0];
    const float* shift   = (const float*)d_in[1];
    const float* wkv0    = (const float*)d_in[2];
    const float* v_first = (const float*)d_in[3];
    const float* x_r = (const float*)d_in[4];
    const float* x_w = (const float*)d_in[5];
    const float* x_k = (const float*)d_in[6];
    const float* x_v = (const float*)d_in[7];
    const float* x_a = (const float*)d_in[8];
    const float* x_g = (const float*)d_in[9];
    const float* w0 = (const float*)d_in[10];
    const float* w1 = (const float*)d_in[11];
    const float* w2 = (const float*)d_in[12];
    const float* a0 = (const float*)d_in[13];
    const float* a1 = (const float*)d_in[14];
    const float* a2 = (const float*)d_in[15];
    const float* v0 = (const float*)d_in[16];
    const float* v1 = (const float*)d_in[17];
    const float* v2 = (const float*)d_in[18];
    const float* g1 = (const float*)d_in[19];
    const float* g2 = (const float*)d_in[20];
    const float* k_k = (const float*)d_in[21];
    const float* k_a = (const float*)d_in[22];
    const float* r_k = (const float*)d_in[23];
    const float* W_r = (const float*)d_in[24];
    const float* W_k = (const float*)d_in[25];
    const float* W_v = (const float*)d_in[26];
    const float* W_o = (const float*)d_in[27];
    const float* gn_w = (const float*)d_in[28];
    const float* gn_b = (const float*)d_in[29];

    const size_t E = (size_t)BT * Cn;   // 4,194,304 ( == 2048*2048 too )
    char* ws = (char*)d_ws;
    size_t off = 0;
    auto alloc = [&](size_t bytes) -> void* {
        void* p = ws + off;
        off = (off + bytes + 255) & ~(size_t)255;
        return p;
    };

    __bf16* xr_b = (__bf16*)alloc(E * 2);
    __bf16* xw_b = (__bf16*)alloc(E * 2);
    __bf16* xk_b = (__bf16*)alloc(E * 2);
    __bf16* xv_b = (__bf16*)alloc(E * 2);
    __bf16* xa_b = (__bf16*)alloc(E * 2);
    __bf16* xg_b = (__bf16*)alloc(E * 2);
    __bf16* Wr_b = (__bf16*)alloc(E * 2);
    __bf16* Wk_b = (__bf16*)alloc(E * 2);
    __bf16* Wv_b = (__bf16*)alloc(E * 2);
    __bf16* Wo_b = (__bf16*)alloc(E * 2);
    float* r_f   = (float*)alloc(E * 4);
    float* k_f   = (float*)alloc(E * 4);
    float* v_f   = (float*)alloc(E * 4);
    float* dec_f = (float*)alloc(E * 4);
    float* aa_f  = (float*)alloc(E * 4);
    float* bb_f  = (float*)alloc(E * 4);
    float* g_f   = (float*)alloc(E * 4);
    float* o_f   = (float*)alloc(E * 4);
    __bf16* y_b  = (__bf16*)alloc(E * 2);
    float* h_w   = (float*)alloc((size_t)BT * 64 * 4);
    float* h_a   = (float*)alloc((size_t)BT * 64 * 4);
    float* h_v   = (float*)alloc((size_t)BT * 32 * 4);
    float* h_g   = (float*)alloc((size_t)BT * 128 * 4);

    // 1. weights -> bf16
    int cvtg = (int)(E / 256);
    k_cvt_bf16<<<cvtg, 256, 0, stream>>>(W_r, Wr_b, E);
    k_cvt_bf16<<<cvtg, 256, 0, stream>>>(W_k, Wk_b, E);
    k_cvt_bf16<<<cvtg, 256, 0, stream>>>(W_v, Wv_b, E);
    k_cvt_bf16<<<cvtg, 256, 0, stream>>>(W_o, Wo_b, E);

    // 2. token shift + mixes
    k_mix<<<(int)((E + 255) / 256), 256, 0, stream>>>(hid, shift, x_r, x_w, x_k, x_v, x_a, x_g,
                                                      xr_b, xw_b, xk_b, xv_b, xa_b, xg_b);

    // 3. big WMMA GEMMs: r, k, v
    dim3 gg(Cn / 128, BT / 128);
    k_gemm_bf16<<<gg, 256, 0, stream>>>(xr_b, Wr_b, r_f, BT, Cn, Cn);
    k_gemm_bf16<<<gg, 256, 0, stream>>>(xk_b, Wk_b, k_f, BT, Cn, Cn);
    k_gemm_bf16<<<gg, 256, 0, stream>>>(xv_b, Wv_b, v_f, BT, Cn, Cn);

    // 4. low-rank stage 1
    k_lr1<<<(BT * 64 + 255) / 256, 256, 0, stream>>>(xw_b, w1, h_w, BT, Cn, 64, 1);
    k_lr1<<<(BT * 64 + 255) / 256, 256, 0, stream>>>(xa_b, a1, h_a, BT, Cn, 64, 0);
    k_lr1<<<(BT * 32 + 255) / 256, 256, 0, stream>>>(xv_b, v1, h_v, BT, Cn, 32, 0);
    k_lr1<<<(BT * 128 + 255) / 256, 256, 0, stream>>>(xg_b, g1, h_g, BT, Cn, 128, 2);

    // 5. fused stage 2 + elementwise prep for scan
    k_fuse<<<(BT * Hh) / 8, 256, 0, stream>>>(h_w, h_a, h_v, h_g, w0, a0, v0,
                                              w2, a2, v2, g2, k_k, k_a, v_first,
                                              k_f, v_f, dec_f, aa_f, bb_f, g_f);

    // 6. WKV7 recurrent scan
    k_scan<<<Bn * Hh, 64, 0, stream>>>(r_f, k_f, v_f, dec_f, aa_f, bb_f, wkv0, o_f);

    // 7. groupnorm + bonus + gate
    k_gn<<<(BT * Hh) / 8, 256, 0, stream>>>(o_f, r_f, k_f, v_f, g_f, gn_w, gn_b, r_k, y_b);

    // 8. output projection
    k_gemm_bf16<<<gg, 256, 0, stream>>>(y_b, Wo_b, (float*)d_out, BT, Cn, Cn);
}
